// Multimodal_ResidualFlow_DiffEmbTransformer_4140348473587
// MI455X (gfx1250) — compile-verified, matrix operand order checked by reading, not run on hardware
//
#include <hip/hip_runtime.h>

#define KNN 20
#define BB  8
#define NP  4096
#define BN  (BB * NP)        // 32768 points total
#define M1  (BN * KNN)       // 655360 rows for layers 1-4

typedef __attribute__((ext_vector_type(16))) __bf16 v16bf;
typedef __attribute__((ext_vector_type(8)))  __bf16 v8bf;
typedef __attribute__((ext_vector_type(8)))  float  v8f;
typedef __attribute__((ext_vector_type(4)))  int    v4i;

typedef __attribute__((address_space(1))) v4i* g_v4i_ptr;   // global int4*
typedef __attribute__((address_space(3))) v4i* l_v4i_ptr;   // LDS int4*

#if __has_builtin(__builtin_amdgcn_global_load_async_to_lds_b128) && \
    __has_builtin(__builtin_amdgcn_s_wait_asynccnt)
#define USE_ASYNC_LDS 1
#else
#define USE_ASYNC_LDS 0
#endif

static __device__ __forceinline__ __bf16 f2b(float f) { return (__bf16)f; }
static __device__ __forceinline__ float  b2f(__bf16 h) { return (float)h; }

// ---------------------------------------------------------------------------
// Weight prep: W (Cin x Cout, fp32) -> WT (Cout x KdPad, bf16), K zero-padded
// ---------------------------------------------------------------------------
__global__ __launch_bounds__(256) void prep_w_kernel(const float* __restrict__ W,
                                                     __bf16* __restrict__ WT,
                                                     int Cin, int Cout, int KdPad) {
    int i = blockIdx.x * blockDim.x + threadIdx.x;
    if (i >= Cout * KdPad) return;
    int n = i / KdPad, k = i % KdPad;
    WT[i] = (k < Cin) ? f2b(W[(size_t)k * Cout + n]) : f2b(0.0f);
}

__global__ __launch_bounds__(256) void zero_kernel(float* __restrict__ p, int n) {
    int i = blockIdx.x * blockDim.x + threadIdx.x;
    if (i < n) p[i] = 0.0f;
}

// ---------------------------------------------------------------------------
// KNN: one wave32 per query point. Per-lane sorted top-20 of a strided subset,
// merged across lanes in LDS with butterfly shfl_xor min-selection.
// Emits neighbor indices in ascending distance (self first), like top_k.
// ---------------------------------------------------------------------------
__global__ __launch_bounds__(128) void knn_kernel(const float* __restrict__ x,
                                                  int* __restrict__ idx) {
    __shared__ float sd[4][32 * KNN];
    __shared__ int   si[4][32 * KNN];
    const int wave = threadIdx.x >> 5;
    const int lane = threadIdx.x & 31;
    const int gw = blockIdx.x * 4 + wave;      // global point id = b*NP + i
    const int b = gw / NP, qi = gw % NP;
    const float* xb = x + (size_t)b * 3 * NP;
    const float qx = xb[qi], qy = xb[NP + qi], qz = xb[2 * NP + qi];

    float d[KNN]; int id[KNN];
#pragma unroll
    for (int t = 0; t < KNN; ++t) { d[t] = 1e30f; id[t] = 0; }

    for (int j = lane; j < NP; j += 32) {
        float dx = qx - xb[j], dy = qy - xb[NP + j], dz = qz - xb[2 * NP + j];
        float dist = dx * dx + dy * dy + dz * dz;
        if (dist < d[KNN - 1]) {
            d[KNN - 1] = dist; id[KNN - 1] = j;
#pragma unroll
            for (int t = KNN - 1; t > 0; --t) {
                if (d[t] < d[t - 1]) {
                    float td = d[t]; d[t] = d[t - 1]; d[t - 1] = td;
                    int   ti = id[t]; id[t] = id[t - 1]; id[t - 1] = ti;
                }
            }
        }
    }
#pragma unroll
    for (int t = 0; t < KNN; ++t) {
        sd[wave][lane * KNN + t] = d[t];
        si[wave][lane * KNN + t] = id[t];
    }
    __syncthreads();

    float* wd = sd[wave];
    int*   wi = si[wave];
    for (int r = 0; r < KNN; ++r) {
        float best = 1e30f; int bslot = -1;
#pragma unroll
        for (int t = 0; t < KNN; ++t) {
            float v = wd[lane * KNN + t];
            if (v < best) { best = v; bslot = lane * KNN + t; }
        }
#pragma unroll
        for (int s = 16; s > 0; s >>= 1) {
            float ob = __shfl_xor(best, s, 32);
            int   os = __shfl_xor(bslot, s, 32);
            if (ob < best || (ob == best && os < bslot)) { best = ob; bslot = os; }
        }
        if (lane == 0) {
            idx[(size_t)gw * KNN + r] = wi[bslot];
            wd[bslot] = 1e30f;
        }
        __syncthreads();
    }
}

// ---------------------------------------------------------------------------
// Gather edge features: row (b,n,k) -> [x[:,j], x[:,n]] zero-padded to 32 ch.
// ---------------------------------------------------------------------------
__global__ __launch_bounds__(256) void gather_kernel(const float* __restrict__ x,
                                                     const int* __restrict__ idx,
                                                     __bf16* __restrict__ f) {
    int row = blockIdx.x * blockDim.x + threadIdx.x;
    if (row >= M1) return;
    int bn = row / KNN;
    int n = bn % NP, b = bn / NP;
    int j = idx[row];
    const float* xb = x + (size_t)b * 3 * NP;

    union { __bf16 o[32]; uint4 u[4]; } r;
#pragma unroll
    for (int t = 0; t < 32; ++t) r.o[t] = f2b(0.0f);
    r.o[0] = f2b(xb[j]);      r.o[1] = f2b(xb[NP + j]); r.o[2] = f2b(xb[2 * NP + j]);
    r.o[3] = f2b(xb[n]);      r.o[4] = f2b(xb[NP + n]); r.o[5] = f2b(xb[2 * NP + n]);

    uint4* dst = (uint4*)(f + (size_t)row * 32);
    dst[0] = r.u[0]; dst[1] = r.u[1]; dst[2] = r.u[2]; dst[3] = r.u[3];
}

// ---------------------------------------------------------------------------
// WMMA GEMM: C[M,N] = A[M,Kd] * BT[N,Kd]^T  (bf16 in, fp32 accum, bf16 out)
// Block = 4 waves computing 64 rows x 64 cols. The shared 64-col B panel is
// staged into LDS in KC=128 super-chunks via async global->LDS loads
// (ASYNCcnt path), then consumed with conflict-free ds_load_b128 reads
// (row pad of 8 elems -> 17 bank-quad stride, coprime with 16).
// ---------------------------------------------------------------------------
#define GEMM_KC  128
#define GEMM_LDB (GEMM_KC + 8)

__global__ __launch_bounds__(128) void gemm_bf16_kernel(const __bf16* __restrict__ A,
                                                        const __bf16* __restrict__ BT,
                                                        __bf16* __restrict__ C,
                                                        int M, int N, int Kd) {
    extern __shared__ __bf16 ldsB[];          // 64 * GEMM_LDB elements
    const int lane = threadIdx.x & 31;
    const int wave = threadIdx.x >> 5;
    const int m0   = (blockIdx.x * 4 + wave) * 16;
    const int n0   = blockIdx.y * 64;
    const int half = lane >> 4;               // 0: K{0..7,16..23}; 1: K{8..15,24..31}
    const int lrow = lane & 15;
    const int koff = half * 8;

    v8f acc[4] = {};
    const __bf16* arow = A + (size_t)(m0 + lrow) * Kd;

    for (int kc0 = 0; kc0 < Kd; kc0 += GEMM_KC) {
        const int kcEff = (Kd - kc0 < GEMM_KC) ? (Kd - kc0) : GEMM_KC;
        const int cpr   = kcEff >> 3;         // 16-byte chunks per panel row
        for (int c = threadIdx.x; c < 64 * cpr; c += 128) {
            int row = c / cpr;
            int kk  = (c - row * cpr) << 3;
            const __bf16* src = BT + (size_t)(n0 + row) * Kd + kc0 + kk;
            __bf16*       dst = ldsB + row * GEMM_LDB + kk;
#if USE_ASYNC_LDS
            __builtin_amdgcn_global_load_async_to_lds_b128((g_v4i_ptr)src, (l_v4i_ptr)dst, 0, 0);
#else
            *(v8bf*)dst = *(const v8bf*)src;
#endif
        }
#if USE_ASYNC_LDS
        __builtin_amdgcn_s_wait_asynccnt(0);
#endif
        __syncthreads();

        for (int kc = 0; kc < kcEff; kc += 32) {
            union { v16bf v; v8bf h[2]; } af;
            af.h[0] = *(const v8bf*)(arow + kc0 + kc + koff);
            af.h[1] = *(const v8bf*)(arow + kc0 + kc + koff + 16);
#pragma unroll
            for (int t = 0; t < 4; ++t) {
                const __bf16* brow = ldsB + (t * 16 + lrow) * GEMM_LDB + kc;
                union { v16bf v; v8bf h[2]; } bf_;
                bf_.h[0] = *(const v8bf*)(brow + koff);
                bf_.h[1] = *(const v8bf*)(brow + koff + 16);
                acc[t] = __builtin_amdgcn_wmma_f32_16x16x32_bf16(
                    false, af.v, false, bf_.v, (short)0, acc[t], false, false);
            }
        }
        __syncthreads();
    }
#pragma unroll
    for (int t = 0; t < 4; ++t) {
#pragma unroll
        for (int r = 0; r < 8; ++r) {
            int row = m0 + r + half * 8;
            int col = n0 + t * 16 + lrow;
            C[(size_t)row * N + col] = f2b(acc[t][r]);
        }
    }
}

// ---------------------------------------------------------------------------
// Channel-wise sum / sum^2 over all M rows (for batch-norm statistics).
// ---------------------------------------------------------------------------
__global__ __launch_bounds__(256) void stats_kernel(const __bf16* __restrict__ t,
                                                    float* __restrict__ sums,
                                                    long long Mrows, int N) {
    extern __shared__ float ls[];   // 2*N floats
    for (int i = threadIdx.x; i < 2 * N; i += blockDim.x) ls[i] = 0.0f;
    __syncthreads();
    long long total = Mrows * (long long)N;
    long long stride = (long long)gridDim.x * blockDim.x;
    for (long long i = (long long)blockIdx.x * blockDim.x + threadIdx.x; i < total; i += stride) {
        float v = b2f(t[i]);
        int c = (int)(i % N);
        atomicAdd(&ls[c], v);
        atomicAdd(&ls[N + c], v * v);
    }
    __syncthreads();
    for (int c = threadIdx.x; c < N; c += blockDim.x) {
        atomicAdd(&sums[c], ls[c]);
        atomicAdd(&sums[N + c], ls[N + c]);
    }
}

__global__ __launch_bounds__(512) void finalize_stats_kernel(const float* __restrict__ sums,
                                                             const float* __restrict__ g,
                                                             const float* __restrict__ bb,
                                                             float* __restrict__ ss,
                                                             float cnt_inv, int N) {
    int c = blockIdx.x * blockDim.x + threadIdx.x;
    if (c >= N) return;
    float mean = sums[c] * cnt_inv;
    float var  = sums[N + c] * cnt_inv - mean * mean;
    float sc   = g[c] * rsqrtf(var + 1e-5f);
    ss[c]      = sc;
    ss[N + c]  = bb[c] - mean * sc;
}

// ---------------------------------------------------------------------------
// Apply BN+ReLU to pre-activation tensor; write next-layer activations (opt)
// and max over the 20 neighbor rows into the concat buffer at col_off.
// ---------------------------------------------------------------------------
__global__ __launch_bounds__(256) void apply_kernel(const __bf16* __restrict__ t,
                                                    const float* __restrict__ ss,
                                                    __bf16* __restrict__ aout,
                                                    __bf16* __restrict__ cat,
                                                    int col_off, int C) {
    int bn = blockIdx.x;
    int c  = threadIdx.x;
    float sc = ss[c], sh = ss[C + c];
    float m = 0.0f;   // post-ReLU values are >= 0
    const __bf16* trow = t + (size_t)bn * KNN * C + c;
    __bf16* arow = aout ? (aout + (size_t)bn * KNN * C + c) : nullptr;
    for (int k = 0; k < KNN; ++k) {
        float v = fmaf(sc, b2f(trow[(size_t)k * C]), sh);
        v = v > 0.0f ? v : 0.0f;
        if (arow) arow[(size_t)k * C] = f2b(v);
        m = v > m ? v : m;
    }
    cat[(size_t)bn * 512 + col_off + c] = f2b(m);
}

// ---------------------------------------------------------------------------
// Final layer epilogue: BN+ReLU + transposed fp32 store (B, 512, N).
// ---------------------------------------------------------------------------
__global__ __launch_bounds__(512) void final_kernel(const __bf16* __restrict__ t5,
                                                    const float* __restrict__ ss,
                                                    float* __restrict__ out) {
    int bn = blockIdx.x;
    int c  = threadIdx.x;
    int b = bn / NP, n = bn % NP;
    float v = fmaf(ss[c], b2f(t5[(size_t)bn * 512 + c]), ss[512 + c]);
    v = v > 0.0f ? v : 0.0f;
    out[((size_t)b * 512 + c) * NP + n] = v;
}

// ---------------------------------------------------------------------------
extern "C" void kernel_launch(void* const* d_in, const int* in_sizes, int n_in,
                              void* d_out, int out_size, void* d_ws, size_t ws_size,
                              hipStream_t stream) {
    (void)in_sizes; (void)n_in; (void)out_size; (void)ws_size;
    const float* x  = (const float*)d_in[0];
    const float* W1 = (const float*)d_in[1];
    const float* W2 = (const float*)d_in[2];
    const float* W3 = (const float*)d_in[3];
    const float* W4 = (const float*)d_in[4];
    const float* W5 = (const float*)d_in[5];
    const float* g1 = (const float*)d_in[6];  const float* b1 = (const float*)d_in[7];
    const float* g2 = (const float*)d_in[8];  const float* b2 = (const float*)d_in[9];
    const float* g3 = (const float*)d_in[10]; const float* b3 = (const float*)d_in[11];
    const float* g4 = (const float*)d_in[12]; const float* b4 = (const float*)d_in[13];
    const float* g5 = (const float*)d_in[14]; const float* b5 = (const float*)d_in[15];

    char* ws = (char*)d_ws;
    size_t off = 0;
    auto alloc = [&](size_t bytes) -> void* {
        void* p = (void*)(ws + off);
        off += (bytes + 255) & ~(size_t)255;
        return p;
    };

    int*    idxb = (int*)   alloc((size_t)M1 * sizeof(int));
    __bf16* bufA = (__bf16*)alloc((size_t)M1 * 64  * 2);   // f(32ch), later a2(64)
    __bf16* bufB = (__bf16*)alloc((size_t)M1 * 128 * 2);   // a1(64),  later a3(128)
    __bf16* tbuf = (__bf16*)alloc((size_t)M1 * 256 * 2);   // reused for all pre-BN tensors
    __bf16* catb = (__bf16*)alloc((size_t)BN * 512 * 2);
    __bf16* WT1  = (__bf16*)alloc((size_t)64  * 32  * 2);
    __bf16* WT2  = (__bf16*)alloc((size_t)64  * 64  * 2);
    __bf16* WT3  = (__bf16*)alloc((size_t)128 * 64  * 2);
    __bf16* WT4  = (__bf16*)alloc((size_t)256 * 128 * 2);
    __bf16* WT5  = (__bf16*)alloc((size_t)512 * 512 * 2);
    float*  sums = (float*) alloc(2048 * sizeof(float));   // per-layer [sum|sumsq]
    float*  ssb  = (float*) alloc(2048 * sizeof(float));   // per-layer [scale|shift]

    float* sums1 = sums;        float* ss1 = ssb;
    float* sums2 = sums + 128;  float* ss2 = ssb + 128;
    float* sums3 = sums + 256;  float* ss3 = ssb + 256;
    float* sums4 = sums + 512;  float* ss4 = ssb + 512;
    float* sums5 = sums + 1024; float* ss5 = ssb + 1024;

    const size_t gemm_lds = (size_t)64 * GEMM_LDB * sizeof(__bf16);  // 17408 B

    // 1) Weight prep (bf16, transposed, K zero-padded to mult. of 32)
    prep_w_kernel<<<(64 * 32 + 255) / 256, 256, 0, stream>>>(W1, WT1, 6, 64, 32);
    prep_w_kernel<<<(64 * 64 + 255) / 256, 256, 0, stream>>>(W2, WT2, 64, 64, 64);
    prep_w_kernel<<<(128 * 64 + 255) / 256, 256, 0, stream>>>(W3, WT3, 64, 128, 64);
    prep_w_kernel<<<(256 * 128 + 255) / 256, 256, 0, stream>>>(W4, WT4, 128, 256, 128);
    prep_w_kernel<<<(512 * 512 + 255) / 256, 256, 0, stream>>>(W5, WT5, 512, 512, 512);
    zero_kernel<<<(2048 + 255) / 256, 256, 0, stream>>>(sums, 2048);

    // 2) KNN + gather
    knn_kernel<<<BN / 4, 128, 0, stream>>>(x, idxb);
    gather_kernel<<<(M1 + 255) / 256, 256, 0, stream>>>(x, idxb, bufA);

    const float inv1 = 1.0f / (float)M1;
    const float inv5 = 1.0f / (float)BN;

    // Layer 1: (M1 x 64) = f(32) @ WT1
    gemm_bf16_kernel<<<dim3(M1 / 64, 64 / 64), 128, gemm_lds, stream>>>(bufA, WT1, tbuf, M1, 64, 32);
    stats_kernel<<<2048, 256, 2 * 64 * sizeof(float), stream>>>(tbuf, sums1, (long long)M1, 64);
    finalize_stats_kernel<<<1, 64, 0, stream>>>(sums1, g1, b1, ss1, inv1, 64);
    apply_kernel<<<BN, 64, 0, stream>>>(tbuf, ss1, bufB, catb, 0, 64);

    // Layer 2: (M1 x 64) = a1(64) @ WT2
    gemm_bf16_kernel<<<dim3(M1 / 64, 64 / 64), 128, gemm_lds, stream>>>(bufB, WT2, tbuf, M1, 64, 64);
    stats_kernel<<<2048, 256, 2 * 64 * sizeof(float), stream>>>(tbuf, sums2, (long long)M1, 64);
    finalize_stats_kernel<<<1, 64, 0, stream>>>(sums2, g2, b2, ss2, inv1, 64);
    apply_kernel<<<BN, 64, 0, stream>>>(tbuf, ss2, bufA, catb, 64, 64);

    // Layer 3: (M1 x 128) = a2(64) @ WT3
    gemm_bf16_kernel<<<dim3(M1 / 64, 128 / 64), 128, gemm_lds, stream>>>(bufA, WT3, tbuf, M1, 128, 64);
    stats_kernel<<<2048, 256, 2 * 128 * sizeof(float), stream>>>(tbuf, sums3, (long long)M1, 128);
    finalize_stats_kernel<<<1, 128, 0, stream>>>(sums3, g3, b3, ss3, inv1, 128);
    apply_kernel<<<BN, 128, 0, stream>>>(tbuf, ss3, bufB, catb, 128, 128);

    // Layer 4: (M1 x 256) = a3(128) @ WT4 ; only the max feeds forward
    gemm_bf16_kernel<<<dim3(M1 / 64, 256 / 64), 128, gemm_lds, stream>>>(bufB, WT4, tbuf, M1, 256, 128);
    stats_kernel<<<2048, 256, 2 * 256 * sizeof(float), stream>>>(tbuf, sums4, (long long)M1, 256);
    finalize_stats_kernel<<<1, 256, 0, stream>>>(sums4, g4, b4, ss4, inv1, 256);
    apply_kernel<<<BN, 256, 0, stream>>>(tbuf, ss4, nullptr, catb, 256, 256);

    // Layer 5: (BN x 512) = cat(512) @ WT5 ; BN+ReLU + transposed output
    gemm_bf16_kernel<<<dim3(BN / 64, 512 / 64), 128, gemm_lds, stream>>>(catb, WT5, tbuf, BN, 512, 512);
    stats_kernel<<<2048, 256, 2 * 512 * sizeof(float), stream>>>(tbuf, sums5, (long long)BN, 512);
    finalize_stats_kernel<<<1, 512, 0, stream>>>(sums5, g5, b5, ss5, inv5, 512);
    final_kernel<<<BN, 512, 0, stream>>>(tbuf, ss5, (float*)d_out);
}
